// HCCF_42932493091124
// MI455X (gfx1250) — matrix operand order repaired and checked by reading, not compiled, not run on hardware
//
#include <hip/hip_runtime.h>
#include <hip/hip_bf16.h>

// Problem constants (match reference)
#define U_NODES 100000
#define I_NODES 50000
#define N_NODES 150000
#define DIM 64
#define NNZ 2000000            // 2 * NNZ_INTER (symmetric)
#define N_LAYERS 3
#define NE (N_NODES * DIM)     // 9,600,000 elements per [N,64] matrix
#define PAD_ROWS 16            // zero rows so item range rounds up to a multiple of 32
#define NEP (NE + PAD_ROWS * DIM)
#define I_PADDED (I_NODES + PAD_ROWS)   // 50016, multiple of 32
#define CHUNK 2048             // rows per block in lat reduction (multiple of 32)
#define RSTAGE 128             // rows staged in LDS per sync round
#define PSTRIDE (RSTAGE + 2)   // transposed P tile stride (bank-conflict pad)

typedef __bf16 v16bf __attribute__((ext_vector_type(16)));
typedef float  v8f   __attribute__((ext_vector_type(8)));

__device__ __forceinline__ void atomAddF(float* p, float v) {
    unsafeAtomicAdd(p, v);     // global_atomic_add_f32 on gfx1250
}

// ---------------------------------------------------------------------------
// prev = acc = concat(uEmbeds, iEmbeds); hyperBF = bf16(prev); pad rows = 0.
// ---------------------------------------------------------------------------
__global__ void init_embeds(const float* __restrict__ uE,
                            const float* __restrict__ iE,
                            float* __restrict__ prev,
                            float* __restrict__ acc,
                            __bf16* __restrict__ hyperBF) {
    int i = blockIdx.x * 256 + threadIdx.x;
    if (i >= NEP) return;
    float v = 0.0f;
    if (i < NE) {
        int n = i >> 6;
        v = (n < U_NODES) ? uE[i] : iE[i - U_NODES * DIM];
        acc[i] = v;
    }
    prev[i]    = v;
    hyperBF[i] = (__bf16)v;
}

// ---------------------------------------------------------------------------
// tem[rows[e]] += vals[e] * prev[cols[e]]   (segment_sum scatter)
// ---------------------------------------------------------------------------
__global__ void spmm_scatter(const int* __restrict__ rows,
                             const int* __restrict__ cols,
                             const float* __restrict__ vals,
                             const float* __restrict__ prev,
                             float* __restrict__ tem) {
    int t = blockIdx.x * 256 + threadIdx.x;
    int e = t >> 2;
    if (e >= NNZ) return;
    int g = (t & 3) * 16;
    int r = rows[e];
    int c = cols[e];
    float v = vals[e];
    const float4* src = (const float4*)(prev + c * DIM + g);
    float* dst = tem + r * DIM + g;
#pragma unroll
    for (int q = 0; q < 4; ++q) {
        float4 x = src[q];
        atomAddF(dst + q * 4 + 0, v * x.x);
        atomAddF(dst + q * 4 + 1, v * x.y);
        atomAddF(dst + q * 4 + 2, v * x.z);
        atomAddF(dst + q * 4 + 3, v * x.w);
    }
}

// ---------------------------------------------------------------------------
// lat(64x64) = hyper^T @ prev over the user or item row range.
// 512 threads = 16 waves covering the 4x4 grid of 16x16 output tiles.
// H tile staged to LDS via async copy (row-major bf16, reused by all 16 waves).
// P tile staged f32->bf16 TRANSPOSED so B-fragments are contiguous LDS reads.
// All chunk sizes are multiples of 32 (buffers zero-padded), so fragment
// loads are unconditional (no divergent bounds checks).
// ---------------------------------------------------------------------------
__global__ void __launch_bounds__(512)
lat_reduce(const __bf16* __restrict__ hyperBF,
           const float* __restrict__ prev,
           float* __restrict__ latU,
           float* __restrict__ latI) {
    __shared__ __align__(16) __bf16 Hlds[RSTAGE * DIM];    // [r][c], 16 KB
    __shared__ __align__(16) __bf16 PldsT[DIM * PSTRIDE];  // [c][r], 16.25 KB

    const int nUblk = (U_NODES + CHUNK - 1) / CHUNK;       // 49
    int b = blockIdx.x;
    int rbase, rcount;
    const __bf16* hyp;
    const float* p;
    float* lat;
    if (b < nUblk) {
        rbase = b * CHUNK;
        rcount = U_NODES - rbase; if (rcount > CHUNK) rcount = CHUNK;
        hyp = hyperBF; p = prev; lat = latU;
    } else {
        b -= nUblk;
        rbase = b * CHUNK;
        rcount = I_PADDED - rbase; if (rcount > CHUNK) rcount = CHUNK;
        hyp = hyperBF + U_NODES * DIM; p = prev + U_NODES * DIM; lat = latI;
    }
    int wave  = threadIdx.x >> 5;
    int lane  = threadIdx.x & 31;
    int mtile = (wave >> 2) * 16;
    int ntile = (wave & 3) * 16;
    int lh = lane >> 4;     // lane half (0/1)
    int ll = lane & 15;

    unsigned hlds_base = (unsigned)(size_t)(&Hlds[0]);     // LDS byte offset

    v8f c = {};
    for (int s0 = 0; s0 < rcount; s0 += RSTAGE) {
        int nrows = rcount - s0; if (nrows > RSTAGE) nrows = RSTAGE;  // mult of 32

        // --- stage H tile: async global->LDS copy, 8 B per lane per issue ---
        {
            const char* hsrc = (const char*)(hyp + (rbase + s0) * DIM);
            int bytes = nrows * DIM * 2;
            for (int off = threadIdx.x * 8; off < bytes; off += 512 * 8) {
                unsigned lds_addr  = hlds_base + (unsigned)off;
                unsigned long long gaddr = (unsigned long long)(hsrc + off);
                asm volatile("global_load_async_to_lds_b64 %0, %1, off"
                             :: "v"(lds_addr), "v"(gaddr) : "memory");
            }
        }
        // --- stage P tile transposed, f32 -> bf16 ---
        {
            int elems = nrows * DIM;
            for (int i = threadIdx.x; i < elems; i += 512) {
                int r = i >> 6;
                int col = i & 63;
                PldsT[col * PSTRIDE + r] = (__bf16)p[(rbase + s0 + r) * DIM + col];
            }
        }
        asm volatile("s_wait_asynccnt 0x0" ::: "memory");
        __syncthreads();

        for (int ks = 0; ks < nrows; ks += 32) {
            v16bf a, bb;
#pragma unroll
            for (int j = 0; j < 16; ++j) {
                // A[m][k] = H[k][m]: lanes 0-15 K=0-15; lanes 16-31 K=16-31
                int kk = ks + lh * 16 + j;
                a[j] = Hlds[kk * DIM + mtile + ll];
            }
            const __bf16* bp = &PldsT[(ntile + ll) * PSTRIDE + ks + lh * 16];
#pragma unroll
            for (int j = 0; j < 16; ++j) bb[j] = bp[j];   // contiguous 32 B
            c = __builtin_amdgcn_wmma_f32_16x16x32_bf16(false, a, false, bb,
                                                        (short)0, c, false, false);
        }
        __syncthreads();
    }
#pragma unroll
    for (int i = 0; i < 8; ++i) {
        int m = mtile + i + 8 * lh;
        int n = ntile + ll;
        atomAddF(&lat[m * DIM + n], c[i]);
    }
}

// ---------------------------------------------------------------------------
// latT = bf16(transpose(latU)) ++ bf16(transpose(latI))
// Stored as latT[n*64 + k] = lat[k*64 + n] so apply's B-fragments are
// contiguous per-lane loads.
// ---------------------------------------------------------------------------
__global__ void lat_cvt(const float* __restrict__ latU,
                        const float* __restrict__ latI,
                        __bf16* __restrict__ latT) {
    int i = blockIdx.x * 256 + threadIdx.x;   // 0..8191
    if (i >= 2 * DIM * DIM) return;
    int side   = i >> 12;
    int within = i & (DIM * DIM - 1);
    int k = within >> 6;
    int n = within & 63;
    const float* src = side ? latI : latU;
    latT[side * DIM * DIM + n * DIM + k] = (__bf16)src[within];
}

// ---------------------------------------------------------------------------
// next = tem + hyper @ lat   (in place in tem), acc += next.
// One 16-row tile per block; 4 waves cover the 4 column tiles.
// ---------------------------------------------------------------------------
__global__ void __launch_bounds__(128)
apply_hgnn(const __bf16* __restrict__ hyperBF,
           const __bf16* __restrict__ latT,
           float* __restrict__ tem_next,
           float* __restrict__ acc) {
    int r0 = blockIdx.x * 16;
    const __bf16* lt = latT + ((r0 < U_NODES) ? 0 : (DIM * DIM));
    int wave  = threadIdx.x >> 5;
    int lane  = threadIdx.x & 31;
    int ntile = wave * 16;
    int lh = lane >> 4;
    int ll = lane & 15;
    int row = r0 + ll;

    v8f c = {};
#pragma unroll
    for (int ktile = 0; ktile < DIM; ktile += 32) {
        v16bf a, bfrag;
        // A row-major: lane (m=ll) reads K = kbase..kbase+15 (kbase = ktile + 16*lh)
        const __bf16* ap = hyperBF + row * DIM + ktile + lh * 16;
#pragma unroll
        for (int j = 0; j < 8; ++j) a[j]     = ap[j];
#pragma unroll
        for (int j = 0; j < 8; ++j) a[8 + j] = ap[8 + j];
        // B[k][n] = lat[k][n] = latT[n*64 + k]: contiguous 32 B per lane
        const __bf16* bp = lt + (ntile + ll) * DIM + ktile + lh * 16;
#pragma unroll
        for (int j = 0; j < 16; ++j) bfrag[j] = bp[j];
        c = __builtin_amdgcn_wmma_f32_16x16x32_bf16(false, a, false, bfrag,
                                                    (short)0, c, false, false);
    }
#pragma unroll
    for (int i = 0; i < 8; ++i) {
        int m   = i + 8 * lh;
        int idx = (r0 + m) * DIM + ntile + ll;
        float nv = tem_next[idx] + c[i];
        tem_next[idx] = nv;   // becomes next layer's prev
        acc[idx] += nv;
    }
}

// ---------------------------------------------------------------------------
extern "C" void kernel_launch(void* const* d_in, const int* in_sizes, int n_in,
                              void* d_out, int out_size, void* d_ws, size_t ws_size,
                              hipStream_t stream) {
    const float* uE   = (const float*)d_in[0];
    const float* iE   = (const float*)d_in[1];
    const int*   rows = (const int*)d_in[2];
    const int*   cols = (const int*)d_in[3];
    const float* vals = (const float*)d_in[4];
    float* acc = (float*)d_out;

    char* ws = (char*)d_ws;
    float*  bufA    = (float*)ws;                                 // prev (padded)
    float*  bufB    = (float*)(ws + (size_t)NEP * 4);             // tem / next (padded)
    __bf16* hyperBF = (__bf16*)(ws + (size_t)NEP * 8);            // bf16 embeds (padded)
    float*  latU    = (float*)(ws + (size_t)NEP * 8 + (size_t)NEP * 2);
    float*  latI    = latU + DIM * DIM;
    __bf16* latT    = (__bf16*)(latI + DIM * DIM);

    init_embeds<<<(NEP + 255) / 256, 256, 0, stream>>>(uE, iE, bufA, acc, hyperBF);

    const int nLatBlocks = (U_NODES + CHUNK - 1) / CHUNK + (I_PADDED + CHUNK - 1) / CHUNK;
    float* prev = bufA;
    float* tem  = bufB;
    for (int l = 0; l < N_LAYERS; ++l) {
        hipMemsetAsync(tem, 0, (size_t)NEP * 4, stream);
        hipMemsetAsync(latU, 0, (size_t)2 * DIM * DIM * 4, stream);
        spmm_scatter<<<(NNZ * 4) / 256, 256, 0, stream>>>(rows, cols, vals, prev, tem);
        lat_reduce<<<nLatBlocks, 512, 0, stream>>>(hyperBF, prev, latU, latI);
        lat_cvt<<<32, 256, 0, stream>>>(latU, latI, latT);
        apply_hgnn<<<N_NODES / 16, 128, 0, stream>>>(hyperBF, latT, tem, acc);
        float* t = prev; prev = tem; tem = t;
    }
}
